// HANLayer_24575802867876
// MI455X (gfx1250) — matrix operand order repaired and checked by reading, not compiled, not run on hardware
//
#include <hip/hip_runtime.h>

// ---------------- problem constants (match reference) ----------------
static constexpr int NN   = 100000;  // nodes
static constexpr int PP   = 3;       // metapaths
static constexpr int EE   = 800000;  // edges per metapath
static constexpr int INF_ = 128;     // in_size
static constexpr int HH   = 4;       // heads
static constexpr int DD   = 32;      // per-head dim
static constexpr int HD   = HH * DD; // 128
static constexpr float NEG_SLOPE = 0.2f;

typedef __attribute__((ext_vector_type(16))) __bf16 v16bf;
typedef __attribute__((ext_vector_type(8)))  float  v8f;

// ---------------- helpers ----------------
__device__ __forceinline__ float lrelu(float v) {
    return v > 0.0f ? v : NEG_SLOPE * v;
}

// Sign-magnitude monotone float max using integer atomics (works for mixed
// signs when initialized to -inf): positives via signed max, negatives via
// unsigned min.
__device__ __forceinline__ void atomicMaxFloat(float* addr, float val) {
    if (val >= 0.0f)
        atomicMax(reinterpret_cast<int*>(addr), __float_as_int(val));
    else
        atomicMin(reinterpret_cast<unsigned int*>(addr), __float_as_uint(val));
}

// ---------------- kernel 1: feat = h @ W_p  (bf16 WMMA, f32 acc) -------------
// Block: 256 threads (8 waves). Each wave owns one 16-wide column tile of the
// 128 outputs. Block covers 64 rows (4 row-tiles of 16) so the 32 KB W_p LDS
// staging is amortized 4x. W is staged TRANSPOSED ([n][k], padded stride 136)
// so B fragments are k-contiguous -> ds_load_b128. B fragments hoisted to
// registers (4 x v16bf). Full blocks take a branchless store path.
static constexpr int LSTR = 136;     // padded LDS row stride (elements)

__global__ void proj_wmma_kernel(const float* __restrict__ hmat,
                                 const float* __restrict__ Wp,
                                 float* __restrict__ feat, int n_rows) {
    __shared__ __bf16 sA[64 * LSTR];    // 17408 B : A rows   [r][k]
    __shared__ __bf16 sB[128 * LSTR];   // 34816 B : W_p^T    [n][k]

    const int  tid  = threadIdx.x;
    const int  row0 = blockIdx.x * 64;
    const bool full = (row0 + 64) <= n_rows;

    // stage W transposed: sB[n*LSTR + k] = Wp[k*128 + n]  (global read coalesced)
    for (int i = tid; i < 128 * 128; i += 256) {
        int k = i >> 7, n = i & 127;
        sB[n * LSTR + k] = (__bf16)Wp[i];
    }
    // stage A tile: sA[r*LSTR + k] = h[(row0+r)*128 + k]
    if (full) {
        for (int i = tid; i < 64 * 128; i += 256) {
            int r = i >> 7, k = i & 127;
            sA[r * LSTR + k] = (__bf16)hmat[(row0 + r) * 128 + k];
        }
    } else {
        for (int i = tid; i < 64 * 128; i += 256) {
            int r = i >> 7, k = i & 127;
            int gr = row0 + r;
            sA[r * LSTR + k] = (gr < n_rows) ? (__bf16)hmat[gr * 128 + k]
                                             : (__bf16)0.0f;
        }
    }
    __syncthreads();

    const int lane = tid & 31;
    const int wv   = tid >> 5;     // column tile 0..7
    const int g    = lane >> 4;    // lane group
    const int mn   = lane & 15;    // M (for A) / N (for B,C,D)

    // B 32x16 bf16 fragment, k order chosen by this code (A matches): lanes0-15 hold K=0..15, lanes16-31 K=16..31
    // -> per lane 16 contiguous K values starting at kb + g*16.
    v16bf bfrag[4];
#pragma unroll
    for (int s = 0; s < 4; ++s) {
        const int kb = s * 32;
#pragma unroll
        for (int j = 0; j < 16; ++j)
            bfrag[s][j] = sB[(wv * 16 + mn) * LSTR + kb + g * 16 + j];
    }

#pragma unroll
    for (int t = 0; t < 4; ++t) {           // 4 row-tiles of 16
        v8f acc = {0.f, 0.f, 0.f, 0.f, 0.f, 0.f, 0.f, 0.f};
#pragma unroll
        for (int s = 0; s < 4; ++s) {       // K = 128 in 4 steps of 32
            const int kb = s * 32;
            v16bf a;
            // A 16x32 layout (matches B): lanes0-15 K=0..15;
            // lanes16-31 K=16..31 (same M rows).
#pragma unroll
            for (int j = 0; j < 8; ++j)
                a[j]     = sA[(t * 16 + mn) * LSTR + kb + g * 16 + j];
#pragma unroll
            for (int j = 0; j < 8; ++j)
                a[8 + j] = sA[(t * 16 + mn) * LSTR + kb + g * 16 + 8 + j];

            acc = __builtin_amdgcn_wmma_f32_16x16x32_bf16(
                false, a, false, bfrag[s], (short)0, acc, false, false);
        }
        // C/D layout: VGPR r -> M = r + 8*g, N = lane&15.
        const int rbase = row0 + t * 16 + 8 * g;
        if (full) {
#pragma unroll
            for (int r = 0; r < 8; ++r)
                feat[(rbase + r) * 128 + wv * 16 + mn] = acc[r];
        } else {
#pragma unroll
            for (int r = 0; r < 8; ++r) {
                int gr = rbase + r;
                if (gr < n_rows) feat[gr * 128 + wv * 16 + mn] = acc[r];
            }
        }
    }
}

// ---------------- kernel 2: el/er = reduce(feat * attn) ----------------------
__global__ void el_er_kernel(const float* __restrict__ feat,
                             const float* __restrict__ al,
                             const float* __restrict__ ar,
                             float* __restrict__ el, float* __restrict__ er,
                             int nh) {
    int idx = blockIdx.x * blockDim.x + threadIdx.x;
    if (idx >= nh) return;
    int n = idx >> 2, h = idx & 3;
    const float* f = feat + n * 128 + h * 32;
    const float* a = al + h * 32;
    const float* b = ar + h * 32;
    float sl = 0.f, sr = 0.f;
#pragma unroll
    for (int d = 0; d < 32; ++d) {
        float v = f[d];
        sl += v * a[d];
        sr += v * b[d];
    }
    el[idx] = sl;
    er[idx] = sr;
}

// ---------------- init kernels ----------------
__global__ void init_nh_kernel(float* emax, float* denom, int nh) {
    int idx = blockIdx.x * blockDim.x + threadIdx.x;
    if (idx >= nh) return;
    emax[idx]  = -__builtin_inff();
    denom[idx] = 0.0f;
}

__global__ void init_z_kernel(float* z, const float* __restrict__ biasp, int total) {
    int idx = blockIdx.x * blockDim.x + threadIdx.x;
    if (idx < total) z[idx] = biasp[idx & 127];
}

__global__ void init_score_kernel(float* score) {
    if (threadIdx.x < PP) score[threadIdx.x] = 0.0f;
}

// ---------------- kernel 3: segment max of edge logits -----------------------
__global__ void edge_max_kernel(const long long* __restrict__ src,
                                const long long* __restrict__ dst,
                                const float* __restrict__ el,
                                const float* __restrict__ er,
                                float* __restrict__ emax, int eh) {
    int idx = blockIdx.x * blockDim.x + threadIdx.x;
    if (idx >= eh) return;
    int e = idx >> 2, h = idx & 3;
    int s = (int)src[e], d = (int)dst[e];
    float v = lrelu(el[s * 4 + h] + er[d * 4 + h]);
    atomicMaxFloat(&emax[d * 4 + h], v);
}

// ---------------- kernel 4: segment sum of exp -------------------------------
__global__ void edge_denom_kernel(const long long* __restrict__ src,
                                  const long long* __restrict__ dst,
                                  const float* __restrict__ el,
                                  const float* __restrict__ er,
                                  const float* __restrict__ emax,
                                  float* __restrict__ denom, int eh) {
    int idx = blockIdx.x * blockDim.x + threadIdx.x;
    if (idx >= eh) return;
    int e = idx >> 2, h = idx & 3;
    int s = (int)src[e], d = (int)dst[e];
    float v  = lrelu(el[s * 4 + h] + er[d * 4 + h]);
    float em = emax[d * 4 + h];
    em = (em == -__builtin_inff()) ? 0.0f : em;   // zero-in-degree guard
    atomicAdd(&denom[d * 4 + h], expf(v - em));
}

// ---------------- kernel 5: weighted message scatter -------------------------
// One wave per edge; each lane handles a float4 chunk of the 128-wide message.
__global__ void edge_scatter_kernel(const long long* __restrict__ src,
                                    const long long* __restrict__ dst,
                                    const float* __restrict__ el,
                                    const float* __restrict__ er,
                                    const float* __restrict__ emax,
                                    const float* __restrict__ denom,
                                    const float* __restrict__ feat,
                                    float* __restrict__ z, int n_edges) {
    const int lane = threadIdx.x & 31;
    const int eid  = blockIdx.x * 8 + (threadIdx.x >> 5);
    if (eid >= n_edges) return;
    const int s = (int)src[eid], d = (int)dst[eid];
    const int h = lane >> 3;                 // (lane*4)/32

    float v  = lrelu(el[s * 4 + h] + er[d * 4 + h]);
    float em = emax[d * 4 + h];
    em = (em == -__builtin_inff()) ? 0.0f : em;
    float ex    = expf(v - em);
    float den   = fmaxf(denom[d * 4 + h], 1e-9f);
    float alpha = ex / den;

    float4 f = reinterpret_cast<const float4*>(feat + (size_t)s * 128)[lane];
    float* zd = z + (size_t)d * 128 + lane * 4;
    atomicAdd(zd + 0, alpha * f.x);
    atomicAdd(zd + 1, alpha * f.y);
    atomicAdd(zd + 2, alpha * f.z);
    atomicAdd(zd + 3, alpha * f.w);
}

// ---------------- kernel 6: semantic attention score partial -----------------
// score[p] += sum_n ( tanh(z_n @ W1 + b1) @ W2 ). W1 (32 KB) staged in LDS.
__global__ void score_kernel(const float* __restrict__ z,
                             const float* __restrict__ w1,
                             const float* __restrict__ b1,
                             const float* __restrict__ w2,
                             float* __restrict__ score_out, int n_rows) {
    __shared__ float sW1[128 * 64];
    __shared__ float sB1[64];
    __shared__ float sW2[64];
    __shared__ float sred[8];

    const int tid = threadIdx.x;
    for (int i = tid; i < 128 * 64; i += 256) sW1[i] = w1[i];
    if (tid < 64) { sB1[tid] = b1[tid]; sW2[tid] = w2[tid]; }
    __syncthreads();

    const int n = blockIdx.x * 256 + tid;
    float s = 0.0f;
    if (n < n_rows) {
        float acc[64];
#pragma unroll
        for (int k = 0; k < 64; ++k) acc[k] = sB1[k];
        const float* zr = z + (size_t)n * 128;
        for (int i = 0; i < 128; ++i) {
            float zi = zr[i];
            const float* wr = sW1 + i * 64;
#pragma unroll
            for (int k = 0; k < 64; ++k) acc[k] += zi * wr[k];
        }
#pragma unroll
        for (int k = 0; k < 64; ++k) s += tanhf(acc[k]) * sW2[k];
    }
    // wave32 reduce, then block reduce
    for (int off = 16; off > 0; off >>= 1) s += __shfl_down(s, off, 32);
    if ((tid & 31) == 0) sred[tid >> 5] = s;
    __syncthreads();
    if (tid == 0) {
        float t = 0.f;
#pragma unroll
        for (int w = 0; w < 8; ++w) t += sred[w];
        atomicAdd(score_out, t);
    }
}

// ---------------- kernel 7: beta = softmax(score / N) ------------------------
__global__ void beta_kernel(const float* __restrict__ score,
                            float* __restrict__ beta, float invN) {
    float w0 = score[0] * invN, w1 = score[1] * invN, w2 = score[2] * invN;
    float m  = fmaxf(w0, fmaxf(w1, w2));
    float e0 = expf(w0 - m), e1 = expf(w1 - m), e2 = expf(w2 - m);
    float sm = e0 + e1 + e2;
    beta[0] = e0 / sm; beta[1] = e1 / sm; beta[2] = e2 / sm;
}

// ---------------- kernel 8: out = sum_p beta_p * z_p -------------------------
__global__ void combine_kernel(const float* __restrict__ z,
                               const float* __restrict__ beta,
                               float* __restrict__ out, int total) {
    int idx = blockIdx.x * blockDim.x + threadIdx.x;
    if (idx >= total) return;
    float b0 = beta[0], b1 = beta[1], b2 = beta[2];
    out[idx] = b0 * z[idx] + b1 * z[idx + (size_t)total] +
               b2 * z[idx + 2 * (size_t)total];
}

// =============================== launch ======================================
extern "C" void kernel_launch(void* const* d_in, const int* in_sizes, int n_in,
                              void* d_out, int out_size, void* d_ws, size_t ws_size,
                              hipStream_t stream) {
    const float*     h    = (const float*)d_in[0];
    const long long* esrc = (const long long*)d_in[1];   // int64 indices
    const long long* edst = (const long long*)d_in[2];
    const float*     W    = (const float*)d_in[3];       // [P,128,128]
    const float*     al   = (const float*)d_in[4];       // [P,H,D]
    const float*     ar   = (const float*)d_in[5];
    const float*     bias = (const float*)d_in[6];       // [P,128]
    const float*     w1   = (const float*)d_in[7];       // [128,64]
    const float*     b1   = (const float*)d_in[8];       // [64]
    const float*     w2   = (const float*)d_in[9];       // [64,1]
    float*           out  = (float*)d_out;

    // workspace layout (floats)
    float* ws    = (float*)d_ws;
    float* z     = ws;                                   // 3*N*128
    float* feat  = z + (size_t)PP * NN * HD;             // N*128 (reused per path)
    float* el    = feat + (size_t)NN * HD;               // N*H
    float* er    = el + (size_t)NN * HH;
    float* emax  = er + (size_t)NN * HH;
    float* denom = emax + (size_t)NN * HH;
    float* score = denom + (size_t)NN * HH;              // P
    float* beta  = score + PP;                           // P

    const int NH  = NN * HH;       // 400000
    const int EH  = EE * HH;       // 3200000
    const int TOT = NN * HD;       // 12800000

    init_score_kernel<<<1, 32, 0, stream>>>(score);

    for (int p = 0; p < PP; ++p) {
        float* zp = z + (size_t)p * NN * HD;
        proj_wmma_kernel<<<dim3((NN + 63) / 64), 256, 0, stream>>>(
            h, W + (size_t)p * INF_ * HD, feat, NN);
        el_er_kernel<<<(NH + 255) / 256, 256, 0, stream>>>(
            feat, al + p * HH * DD, ar + p * HH * DD, el, er, NH);
        init_nh_kernel<<<(NH + 255) / 256, 256, 0, stream>>>(emax, denom, NH);
        init_z_kernel<<<(TOT + 255) / 256, 256, 0, stream>>>(zp, bias + p * HD, TOT);
        edge_max_kernel<<<(EH + 255) / 256, 256, 0, stream>>>(
            esrc + (size_t)p * EE, edst + (size_t)p * EE, el, er, emax, EH);
        edge_denom_kernel<<<(EH + 255) / 256, 256, 0, stream>>>(
            esrc + (size_t)p * EE, edst + (size_t)p * EE, el, er, emax, denom, EH);
        edge_scatter_kernel<<<(EE + 7) / 8, 256, 0, stream>>>(
            esrc + (size_t)p * EE, edst + (size_t)p * EE, el, er, emax, denom,
            feat, zp, EE);
        score_kernel<<<(NN + 255) / 256, 256, 0, stream>>>(
            zp, w1, b1, w2, score + p, NN);
    }

    beta_kernel<<<1, 1, 0, stream>>>(score, beta, 1.0f / (float)NN);
    combine_kernel<<<(TOT + 255) / 256, 256, 0, stream>>>(z, beta, out, TOT);
}
